// Block_24584392802504
// MI455X (gfx1250) — matrix operand order repaired and checked by reading, not compiled, not run on hardware
//
#include <hip/hip_runtime.h>
#include <hip/hip_bf16.h>
#include <math.h>

// ---------------------------------------------------------------------------
// Transformer block forward for MI455X (gfx1250, wave32, WMMA).
// GEMMs: V_WMMA_F32_16X16X32_BF16, LDS-staged K-panels via
// GLOBAL_LOAD_ASYNC_TO_LDS_B128 (double buffered, ASYNCcnt synchronized).
// Attention is flash-style (no N x N materialization).
// ---------------------------------------------------------------------------

#define DIM      1024
#define HEADS    16
#define HEAD_DIM 64
#define HIDDEN   4096
#define SEQ      1024
#define BATCH    4
#define TOKENS   (BATCH * SEQ)   // 4096

typedef __attribute__((ext_vector_type(16))) __bf16 v16bf;
typedef __attribute__((ext_vector_type(8)))  __bf16 v8bf;
typedef __attribute__((ext_vector_type(8)))  float  v8f;

// fp32 -> bf16, round to nearest even
__device__ __forceinline__ unsigned short f2bf(float f) {
  unsigned int u = __float_as_uint(f);
  unsigned int r = u + 0x7FFFu + ((u >> 16) & 1u);
  return (unsigned short)(r >> 16);
}

__device__ __forceinline__ v8f wmma_bf16(v16bf a, v16bf b, v8f c) {
  return __builtin_amdgcn_wmma_f32_16x16x32_bf16(
      /*neg_a=*/false, a, /*neg_b=*/false, b,
      /*c_mod=*/(short)0, c, /*reuse_a=*/false, /*reuse_b=*/false);
}

// ---- CDNA5 async copy (memory -> LDS), tracked by ASYNCcnt ----------------
// VDST = per-lane LDS byte offset; generic pointers to __shared__ carry the
// wave-relative LDS offset in their low 32 bits (ISA 10.2 aperture mapping).
__device__ __forceinline__ void async_g2l_b128(unsigned int lds_off,
                                               unsigned long long gaddr) {
  asm volatile("global_load_async_to_lds_b128 %0, %1, off"
               :: "v"(lds_off), "v"(gaddr) : "memory");
}
__device__ __forceinline__ void wait_async0() {
  asm volatile("s_wait_asynccnt 0" ::: "memory");
}

// A operand (16x32, M x K), base is row-major [rows x ld], K contiguous.
// lanes 0-15: row=lane,    K = k0+{0..7, 16..23}
// lanes16-31: row=lane-16, K = k0+{8..15,24..31}
__device__ __forceinline__ v16bf load_frag_a(const __bf16* __restrict__ base,
                                             int ld, int row0, int k0) {
  const int lane = threadIdx.x & 31;
  const int row  = row0 + (lane & 15);
  const int kk   = k0 + ((lane >> 4) << 3);
  const __bf16* p = base + (size_t)row * ld + kk;
  v8bf lo = *(const v8bf*)(p);
  v8bf hi = *(const v8bf*)(p + 16);
  v16bf r;
#pragma unroll
  for (int i = 0; i < 8; ++i) { r[i] = lo[i]; r[i + 8] = hi[i]; }
  return r;
}

// B operand (32x16, K x N) taken from NT weight W[col][k] row-major.
// lanes 0-15: col=lane,    K = k0+{0..7, 16..23}
// lanes16-31: col=lane-16, K = k0+{8..15,24..31}
__device__ __forceinline__ v16bf load_frag_b(const __bf16* __restrict__ base,
                                             int ld, int col0, int k0) {
  const int lane = threadIdx.x & 31;
  const int row  = col0 + (lane & 15);
  const int kk   = k0 + ((lane >> 4) << 3);
  const __bf16* p = base + (size_t)row * ld + kk;
  v8bf lo = *(const v8bf*)(p);
  v8bf hi = *(const v8bf*)(p + 16);
  v16bf r;
#pragma unroll
  for (int i = 0; i < 8; ++i) { r[i] = lo[i]; r[i + 8] = hi[i]; }
  return r;
}

// ---------------------------------------------------------------------------
// fp32 -> bf16 bulk cast (4 elems/thread)
// ---------------------------------------------------------------------------
__global__ __launch_bounds__(256)
void cvt_f32_bf16(const float* __restrict__ in, unsigned short* __restrict__ out,
                  int n4) {
  int i = blockIdx.x * blockDim.x + threadIdx.x;
  if (i >= n4) return;
  float4 v = ((const float4*)in)[i];
  unsigned int lo = (unsigned int)f2bf(v.x) | ((unsigned int)f2bf(v.y) << 16);
  unsigned int hi = (unsigned int)f2bf(v.z) | ((unsigned int)f2bf(v.w) << 16);
  ((uint2*)out)[i] = make_uint2(lo, hi);
}

// ---------------------------------------------------------------------------
// LayerNorm: one block (256 thr) per token, DIM=1024, bf16 output
// ---------------------------------------------------------------------------
__global__ __launch_bounds__(256)
void ln_kernel(const float* __restrict__ x, const float* __restrict__ g,
               const float* __restrict__ b, unsigned short* __restrict__ out) {
  const int t   = blockIdx.x;
  const int tid = threadIdx.x;
  const int w    = tid >> 5;
  const int lane = tid & 31;

  float4 v = ((const float4*)(x + (size_t)t * DIM))[tid];
  float s1 = v.x + v.y + v.z + v.w;
  float s2 = v.x * v.x + v.y * v.y + v.z * v.z + v.w * v.w;
#pragma unroll
  for (int off = 16; off > 0; off >>= 1) {
    s1 += __shfl_xor(s1, off, 32);
    s2 += __shfl_xor(s2, off, 32);
  }
  __shared__ float r1[8], r2[8];
  if (lane == 0) { r1[w] = s1; r2[w] = s2; }
  __syncthreads();
  float ts1 = 0.f, ts2 = 0.f;
#pragma unroll
  for (int i = 0; i < 8; ++i) { ts1 += r1[i]; ts2 += r2[i]; }
  const float mean = ts1 * (1.0f / DIM);
  const float var  = ts2 * (1.0f / DIM) - mean * mean;
  const float inv  = rsqrtf(var + 1e-5f);

  float4 gv = ((const float4*)g)[tid];
  float4 bv = ((const float4*)b)[tid];
  float o0 = (v.x - mean) * inv * gv.x + bv.x;
  float o1 = (v.y - mean) * inv * gv.y + bv.y;
  float o2 = (v.z - mean) * inv * gv.z + bv.z;
  float o3 = (v.w - mean) * inv * gv.w + bv.w;
  unsigned int lo = (unsigned int)f2bf(o0) | ((unsigned int)f2bf(o1) << 16);
  unsigned int hi = (unsigned int)f2bf(o2) | ((unsigned int)f2bf(o3) << 16);
  ((uint2*)(out + (size_t)t * DIM))[tid] = make_uint2(lo, hi);
}

// ---------------------------------------------------------------------------
// GEMM: C[M,N] = A_bf16[M,K] @ W_bf16[N,K]^T + bias, fused epilogues.
// Block = 256 thr = 8 waves; block tile 128x128; wave tile 32x64.
// K-panels (128x32 of A and of W) double-buffered in LDS via async copies:
// each thread issues 4x GLOBAL_LOAD_ASYNC_TO_LDS_B128 per K-step, overlapped
// with the 8 WMMAs of the previous step; s_wait_asynccnt + s_barrier sync.
// mode 0: fp32 out + residual    (proj / fc2)
// mode 1: GELU(exact) -> bf16    (fc1)
// mode 2: scatter q (scaled) / k / v^T in bf16 head layouts (qkv)
// ---------------------------------------------------------------------------
#define GEMM_RES  0
#define GEMM_GELU 1
#define GEMM_QKV  2

__global__ __launch_bounds__(256)
void gemm_bf16_nt(const __bf16* __restrict__ A, const __bf16* __restrict__ W,
                  const float* __restrict__ bias,
                  int M, int N, int K, int mode,
                  float* __restrict__ Cf, unsigned short* __restrict__ Cb,
                  const float* __restrict__ resid,
                  unsigned short* __restrict__ q_out,
                  unsigned short* __restrict__ k_out,
                  unsigned short* __restrict__ vt_out) {
  __shared__ unsigned short As[2][128 * 32];   // 2 x 8KB
  __shared__ unsigned short Bs[2][128 * 32];   // 2 x 8KB

  const int tid  = threadIdx.x;
  const int lane = tid & 31;
  const int w    = tid >> 5;
  const int wm = w >> 1;          // 0..3
  const int wn = w & 1;           // 0..1
  const int m0 = wm * 32;         // within block tile
  const int n0 = wn * 64;
  const int mBlk = blockIdx.y * 128;
  const int nBlk = blockIdx.x * 128;

  // async-copy assignment: thread copies 32B of A panel + 32B of B panel.
  const int crow  = tid >> 1;            // 0..127
  const int chalf = (tid & 1) * 16;      // element offset in K
  const unsigned long long gA =
      (unsigned long long)(uintptr_t)(A + (size_t)(mBlk + crow) * K + chalf);
  const unsigned long long gB =
      (unsigned long long)(uintptr_t)(W + (size_t)(nBlk + crow) * K + chalf);
  const unsigned int ldsA = (unsigned int)(uintptr_t)&As[0][crow * 32 + chalf];
  const unsigned int ldsB = (unsigned int)(uintptr_t)&Bs[0][crow * 32 + chalf];
  const unsigned int bufB = 128 * 32 * 2;  // bytes between double buffers

  auto issue = [&](int bi, int k0) {
    unsigned long long a = gA + (unsigned long long)k0 * 2;
    unsigned long long b = gB + (unsigned long long)k0 * 2;
    unsigned int la = ldsA + (unsigned int)bi * bufB;
    unsigned int lb = ldsB + (unsigned int)bi * bufB;
    async_g2l_b128(la,      a);
    async_g2l_b128(la + 16, a + 16);
    async_g2l_b128(lb,      b);
    async_g2l_b128(lb + 16, b + 16);
  };

  v8f acc[2][4];
#pragma unroll
  for (int i = 0; i < 2; ++i)
#pragma unroll
    for (int j = 0; j < 4; ++j) acc[i][j] = (v8f)(0.f);

  const int nk = K >> 5;
  issue(0, 0);
  for (int ik = 0; ik < nk; ++ik) {
    wait_async0();        // this wave's panel loads for buffer ik&1 are in LDS
    __syncthreads();      // everyone's loads done AND everyone done reading
                          // buffer (ik+1)&1 from step ik-1 -> safe to refill
    if (ik + 1 < nk) issue((ik + 1) & 1, (ik + 1) << 5);

    const __bf16* Ab = (const __bf16*)&As[ik & 1][0];
    const __bf16* Bb = (const __bf16*)&Bs[ik & 1][0];
    v16bf a0 = load_frag_a(Ab, 32, m0,      0);
    v16bf a1 = load_frag_a(Ab, 32, m0 + 16, 0);
    v16bf b0 = load_frag_b(Bb, 32, n0,      0);
    v16bf b1 = load_frag_b(Bb, 32, n0 + 16, 0);
    v16bf b2 = load_frag_b(Bb, 32, n0 + 32, 0);
    v16bf b3 = load_frag_b(Bb, 32, n0 + 48, 0);
    acc[0][0] = wmma_bf16(a0, b0, acc[0][0]);
    acc[0][1] = wmma_bf16(a0, b1, acc[0][1]);
    acc[0][2] = wmma_bf16(a0, b2, acc[0][2]);
    acc[0][3] = wmma_bf16(a0, b3, acc[0][3]);
    acc[1][0] = wmma_bf16(a1, b0, acc[1][0]);
    acc[1][1] = wmma_bf16(a1, b1, acc[1][1]);
    acc[1][2] = wmma_bf16(a1, b2, acc[1][2]);
    acc[1][3] = wmma_bf16(a1, b3, acc[1][3]);
  }

  const int hi = lane >> 4, ln = lane & 15;
#pragma unroll
  for (int i = 0; i < 2; ++i) {
#pragma unroll
    for (int j = 0; j < 4; ++j) {
#pragma unroll
      for (int r = 0; r < 8; ++r) {
        const int row = mBlk + m0 + i * 16 + hi * 8 + r;  // C layout: vgpr r -> M=r(+8)
        const int col = nBlk + n0 + j * 16 + ln;
        float v = acc[i][j][r] + bias[col];
        if (mode == GEMM_RES) {
          Cf[(size_t)row * N + col] = v + resid[(size_t)row * N + col];
        } else if (mode == GEMM_GELU) {
          float gl = 0.5f * v * (1.0f + erff(v * 0.70710678118654752f));
          Cb[(size_t)row * N + col] = f2bf(gl);
        } else {  // GEMM_QKV scatter: col in [0,3072)
          const int b  = row >> 10, n = row & 1023;
          const int wh = col >> 10;                  // 0=q 1=k 2=v
          const int h  = (col >> 6) & 15, d = col & 63;
          if (wh == 0) {
            q_out[((((size_t)b * HEADS + h) * SEQ + n) * HEAD_DIM) + d] =
                f2bf(v * 0.125f);                    // HEAD_DIM^-0.5
          } else if (wh == 1) {
            k_out[((((size_t)b * HEADS + h) * SEQ + n) * HEAD_DIM) + d] = f2bf(v);
          } else {                                   // store V transposed [d][token]
            vt_out[((((size_t)b * HEADS + h) * HEAD_DIM + d) * SEQ) + n] = f2bf(v);
          }
        }
      }
    }
  }
}

// ---------------------------------------------------------------------------
// Flash attention: 1 wave = 16 query rows of one (b,h); sweep keys 64-wide.
// S = q @ k^T via WMMA (K=d=64), online softmax, P via 2KB/wave LDS
// (C-layout -> A-layout reshape), O += P @ V^T via WMMA (K=keys=64).
// Writes normalized O as bf16 into att[b][n][h*64+d].
// ---------------------------------------------------------------------------
__global__ __launch_bounds__(256)
void flash_attn(const __bf16* __restrict__ q, const __bf16* __restrict__ k,
                const __bf16* __restrict__ vt, unsigned short* __restrict__ att) {
  __shared__ unsigned short Plds[8][16 * 64];
  const int lane = threadIdx.x & 31;
  const int w    = threadIdx.x >> 5;
  const int gwave = blockIdx.x * 8 + w;
  const int qt = gwave & 63;        // 64 query tiles per (b,h)
  const int bh = gwave >> 6;        // 0..63
  const int q0 = qt * 16;
  const int b  = bh >> 4, h = bh & 15;

  const __bf16* qh = q  + (size_t)bh * SEQ * HEAD_DIM;
  const __bf16* kh = k  + (size_t)bh * SEQ * HEAD_DIM;
  const __bf16* vh = vt + (size_t)bh * HEAD_DIM * SEQ;
  const __bf16* pl = (const __bf16*)&Plds[w][0];

  v16bf qa0 = load_frag_a(qh, HEAD_DIM, q0, 0);
  v16bf qa1 = load_frag_a(qh, HEAD_DIM, q0, 32);

  v8f o[4];
#pragma unroll
  for (int j = 0; j < 4; ++j) o[j] = (v8f)(0.f);
  float mrow[8], lrow[8];
#pragma unroll
  for (int r = 0; r < 8; ++r) { mrow[r] = -INFINITY; lrow[r] = 0.f; }

  const int hi = lane >> 4, ln = lane & 15;

  for (int kb = 0; kb < SEQ; kb += 64) {
    // ---- S tile: 16 x 64 scores ----
    v8f s[4];
#pragma unroll
    for (int j = 0; j < 4; ++j) {
      v16bf b0 = load_frag_b(kh, HEAD_DIM, kb + j * 16, 0);
      v16bf b1 = load_frag_b(kh, HEAD_DIM, kb + j * 16, 32);
      v8f sj = (v8f)(0.f);
      sj = wmma_bf16(qa0, b0, sj);
      sj = wmma_bf16(qa1, b1, sj);
      s[j] = sj;
    }
    // ---- online softmax (rows live across 16-lane half-groups) ----
    float alpha[8];
#pragma unroll
    for (int r = 0; r < 8; ++r) {
      float mx = fmaxf(fmaxf(s[0][r], s[1][r]), fmaxf(s[2][r], s[3][r]));
#pragma unroll
      for (int off = 1; off < 16; off <<= 1) mx = fmaxf(mx, __shfl_xor(mx, off, 16));
      const float mnew = fmaxf(mrow[r], mx);
      alpha[r] = __expf(mrow[r] - mnew);
      float sum = 0.f;
#pragma unroll
      for (int j = 0; j < 4; ++j) {
        float p = __expf(s[j][r] - mnew);
        sum += p;
        Plds[w][(r + 8 * hi) * 64 + j * 16 + ln] = f2bf(p);
      }
#pragma unroll
      for (int off = 1; off < 16; off <<= 1) sum += __shfl_xor(sum, off, 16);
      lrow[r] = lrow[r] * alpha[r] + sum;
      mrow[r] = mnew;
    }
#pragma unroll
    for (int j = 0; j < 4; ++j)
#pragma unroll
      for (int r = 0; r < 8; ++r) o[j][r] *= alpha[r];
    asm volatile("" ::: "memory");   // keep DS stores before DS reads (same wave, HW in-order)
    // ---- O += P @ V^T ----
#pragma unroll
    for (int ks = 0; ks < 2; ++ks) {
      v16bf pa = load_frag_a(pl, 64, 0, ks * 32);
#pragma unroll
      for (int j = 0; j < 4; ++j) {
        v16bf bv = load_frag_b(vh, SEQ, j * 16, kb + ks * 32);
        o[j] = wmma_bf16(pa, bv, o[j]);
      }
    }
  }
  // ---- normalize + write att[b][row][h*64+d] as bf16 ----
#pragma unroll
  for (int j = 0; j < 4; ++j)
#pragma unroll
    for (int r = 0; r < 8; ++r) {
      const int row = q0 + r + 8 * hi;
      const int col = h * HEAD_DIM + j * 16 + ln;
      att[((size_t)(b * SEQ + row)) * DIM + col] = f2bf(o[j][r] / lrow[r]);
    }
}

// ---------------------------------------------------------------------------
// Host-side orchestration
// ---------------------------------------------------------------------------
extern "C" void kernel_launch(void* const* d_in, const int* in_sizes, int n_in,
                              void* d_out, int out_size, void* d_ws, size_t ws_size,
                              hipStream_t stream) {
  const float* x      = (const float*)d_in[0];
  const float* ln1_g  = (const float*)d_in[1];
  const float* ln1_b  = (const float*)d_in[2];
  const float* qkv_w  = (const float*)d_in[3];
  const float* qkv_b  = (const float*)d_in[4];
  const float* proj_w = (const float*)d_in[5];
  const float* proj_b = (const float*)d_in[6];
  const float* ln2_g  = (const float*)d_in[7];
  const float* ln2_b  = (const float*)d_in[8];
  const float* fc1_w  = (const float*)d_in[9];
  const float* fc1_b  = (const float*)d_in[10];
  const float* fc2_w  = (const float*)d_in[11];
  const float* fc2_b  = (const float*)d_in[12];
  float* out = (float*)d_out;

  // workspace carve-up (256B aligned)
  char* base = (char*)d_ws;
  size_t off = 0;
  auto carve = [&](size_t bytes) -> char* {
    char* p = base + off;
    off += (bytes + 255) & ~(size_t)255;
    return p;
  };
  unsigned short* h    = (unsigned short*)carve((size_t)TOKENS * DIM * 2);
  unsigned short* qb   = (unsigned short*)carve((size_t)TOKENS * DIM * 2);
  unsigned short* kb   = (unsigned short*)carve((size_t)TOKENS * DIM * 2);
  unsigned short* vtb  = (unsigned short*)carve((size_t)TOKENS * DIM * 2);
  unsigned short* attb = (unsigned short*)carve((size_t)TOKENS * DIM * 2);
  float*          x1   = (float*)         carve((size_t)TOKENS * DIM * 4);
  unsigned short* gel  = (unsigned short*)carve((size_t)TOKENS * HIDDEN * 2);
  unsigned short* wq   = (unsigned short*)carve((size_t)3 * DIM * DIM * 2);
  unsigned short* wp   = (unsigned short*)carve((size_t)DIM * DIM * 2);
  unsigned short* w1   = (unsigned short*)carve((size_t)HIDDEN * DIM * 2);
  unsigned short* w2   = (unsigned short*)carve((size_t)DIM * HIDDEN * 2);

  auto cvt = [&](const float* src, unsigned short* dst, size_t n) {
    int n4 = (int)(n / 4);
    cvt_f32_bf16<<<(n4 + 255) / 256, 256, 0, stream>>>(src, dst, n4);
  };
  cvt(qkv_w, wq, (size_t)3 * DIM * DIM);
  cvt(proj_w, wp, (size_t)DIM * DIM);
  cvt(fc1_w, w1, (size_t)HIDDEN * DIM);
  cvt(fc2_w, w2, (size_t)DIM * HIDDEN);

  // LN1
  ln_kernel<<<TOKENS, 256, 0, stream>>>(x, ln1_g, ln1_b, h);

  // QKV GEMM: [4096,3072] = h @ qkv_w^T ; scatter q/k/v^T bf16
  gemm_bf16_nt<<<dim3(3 * DIM / 128, TOKENS / 128), 256, 0, stream>>>(
      (const __bf16*)h, (const __bf16*)wq, qkv_b,
      TOKENS, 3 * DIM, DIM, GEMM_QKV,
      nullptr, nullptr, nullptr, qb, kb, vtb);

  // Flash attention: 4096 query-tiles, 8 waves/block
  flash_attn<<<(BATCH * HEADS * (SEQ / 16)) / 8, 256, 0, stream>>>(
      (const __bf16*)qb, (const __bf16*)kb, (const __bf16*)vtb, attb);

  // proj GEMM + residual -> x1 (fp32)
  gemm_bf16_nt<<<dim3(DIM / 128, TOKENS / 128), 256, 0, stream>>>(
      (const __bf16*)attb, (const __bf16*)wp, proj_b,
      TOKENS, DIM, DIM, GEMM_RES,
      x1, nullptr, x, nullptr, nullptr, nullptr);

  // LN2 (reuse h)
  ln_kernel<<<TOKENS, 256, 0, stream>>>(x1, ln2_g, ln2_b, h);

  // FC1 GEMM + exact GELU -> bf16
  gemm_bf16_nt<<<dim3(HIDDEN / 128, TOKENS / 128), 256, 0, stream>>>(
      (const __bf16*)h, (const __bf16*)w1, fc1_b,
      TOKENS, HIDDEN, DIM, GEMM_GELU,
      nullptr, gel, nullptr, nullptr, nullptr, nullptr);

  // FC2 GEMM + residual -> d_out (fp32)
  gemm_bf16_nt<<<dim3(DIM / 128, TOKENS / 128), 256, 0, stream>>>(
      (const __bf16*)gel, (const __bf16*)w2, fc2_b,
      TOKENS, DIM, HIDDEN, GEMM_RES,
      out, nullptr, x1, nullptr, nullptr, nullptr);
}
